// GATNet_20607253086974
// MI455X (gfx1250) — matrix operand order repaired and checked by reading, not compiled, not run on hardware
//
#include <hip/hip_runtime.h>
#include <hip/hip_bf16.h>

// ---------------------------------------------------------------------------
// Problem constants (hardcoded from the reference)
// ---------------------------------------------------------------------------
#define NB    4      // batch
#define NSEC  1024
#define NPRIM 256
#define NN    1536   // total nodes
#define FIN   128
#define FG    128    // HEADS*OUT_F
#define NH    4      // heads
#define OF    32     // out feat per head
#define FC0D  256
#define FC1D  256

typedef _Float16 half_t;
typedef __attribute__((ext_vector_type(16))) _Float16 v16h;
typedef __attribute__((ext_vector_type(2)))  _Float16 v2h;
typedef __attribute__((ext_vector_type(8)))  float    v8f;

__device__ __forceinline__ float lrelu(float t) { return t > 0.f ? t : 0.2f * t; }

// ---------------------------------------------------------------------------
// Elementwise converters
// ---------------------------------------------------------------------------
__global__ void cvt_f32_to_f16_kernel(const float* __restrict__ in,
                                      half_t* __restrict__ out, int n) {
  int t = blockIdx.x * blockDim.x + threadIdx.x;
  if (t < n) out[t] = (half_t)in[t];
}

// Gather sec-node rows of y [B,1536,128] -> hg16 [B*1024,128] f16
__global__ void cvt_sec_kernel(const float* __restrict__ y,
                               half_t* __restrict__ out) {
  int t = blockIdx.x * blockDim.x + threadIdx.x;   // < 4*1024*128
  if (t >= NB * NSEC * FG) return;
  int f = t & 127;
  int i = (t >> 7) & 1023;
  int b = t >> 17;
  out[t] = (half_t)y[((size_t)(b * NN + i) << 7) + f];
}

// Transpose xp [B,1536,128] f32 -> xpT [B,128,1536] f16 (WMMA B operand source)
__global__ void xpT_kernel(const float* __restrict__ xp,
                           half_t* __restrict__ xpT) {
  int t = blockIdx.x * blockDim.x + threadIdx.x;   // < 4*128*1536
  if (t >= NB * FG * NN) return;
  int node = t % NN;
  int f = (t / NN) % FG;
  int b = t / (NN * FG);
  xpT[t] = (half_t)xp[((size_t)(b * NN + node) << 7) + f];
}

// ---------------------------------------------------------------------------
// Weight swizzle into WMMA B-operand per-lane layout (16-bit B 32x16 tile):
//   lane<16 holds N=lane,  K = j      (j=0..15)
//   lane>=16 holds N=lane-16, K = 16+j
// Storage: [kt][nt][lane][j] contiguous -> one v16h load per lane per tile.
// ---------------------------------------------------------------------------
__global__ void swz_weight_kernel(const float* __restrict__ W,
                                  half_t* __restrict__ out,
                                  int KT, int NT) {
  int t = blockIdx.x * blockDim.x + threadIdx.x;
  int total = KT * NT * 512;
  if (t >= total) return;
  int j = t & 15;
  int lane = (t >> 4) & 31;
  int tile = t >> 9;
  int nt = tile % NT;
  int kt = tile / NT;
  int k = kt * 32 + ((lane & 16) ? 16 : 0) + j;
  int n = nt * 16 + (lane & 15);
  out[t] = (half_t)W[(size_t)k * (NT * 16) + n];
}

// ---------------------------------------------------------------------------
// Generic WMMA GEMM: C[M x NT*16] = A[M x KT*32] * Bswz (+bias)(+relu)
// One wave per 16x32 output tile (two adjacent 16-wide N tiles sharing the
// A operand -> two INDEPENDENT accumulator chains; the scheduler interleaves
// the WMMAs into each other's hazard slots instead of v_nops).
// A: row-major f16, per-lane half2 gathers into the 16-bit A layout
// (contiguous K as in Bswz: lane<16: K 0-15; lane>=16: K 16-31) -> coalesces to 2x b128.
// ---------------------------------------------------------------------------
__global__ void gemm_ws_kernel(const half_t* __restrict__ A, int lda,
                               const half_t* __restrict__ Bswz,
                               const float* __restrict__ bias,
                               float* __restrict__ C,
                               half_t* __restrict__ C16,
                               int ldc, int Mtiles, int KT, int NT, int doRelu) {
  int tid = blockIdx.x * blockDim.x + threadIdx.x;
  int w = tid >> 5;
  int lane = threadIdx.x & 31;
  int NP = NT >> 1;                         // N-tile pairs
  int ntp = w % NP;
  int mt = w / NP;
  if (mt >= Mtiles) return;                 // uniform per wave
  int nt0 = 2 * ntp, nt1 = 2 * ntp + 1;
  int ml = lane & 15;
  int hi = lane >> 4;
  v8f acc0 = {0.f, 0.f, 0.f, 0.f, 0.f, 0.f, 0.f, 0.f};
  v8f acc1 = {0.f, 0.f, 0.f, 0.f, 0.f, 0.f, 0.f, 0.f};
  for (int kt = 0; kt < KT; ++kt) {
    const half_t* arow = A + (size_t)(mt * 16 + ml) * lda + kt * 32;
    __builtin_prefetch(arow + 32, 0, 1);
    v16h a;
#pragma unroll
    for (int v = 0; v < 8; ++v) {
      int j = 2 * v;
      int kb = hi * 16 + j;
      v2h pr = *(const v2h*)(arow + kb);
      a[j] = pr[0];
      a[j + 1] = pr[1];
    }
    const half_t* brow = Bswz + ((size_t)(kt * NT) * 32 + lane) * 16;
    v16h b0 = *(const v16h*)(brow + (size_t)nt0 * 512);
    v16h b1 = *(const v16h*)(brow + (size_t)nt1 * 512);
    acc0 = __builtin_amdgcn_wmma_f32_16x16x32_f16(false, a, false, b0,
                                                  (short)0, acc0, false, false);
    acc1 = __builtin_amdgcn_wmma_f32_16x16x32_f16(false, a, false, b1,
                                                  (short)0, acc1, false, false);
  }
  int col0 = nt0 * 16 + ml;
  int col1 = nt1 * 16 + ml;
  float bv0 = bias ? bias[col0] : 0.0f;
  float bv1 = bias ? bias[col1] : 0.0f;
#pragma unroll
  for (int r = 0; r < 8; ++r) {
    int row = mt * 16 + r + hi * 8;
    float v0 = acc0[r] + bv0;
    float v1 = acc1[r] + bv1;
    if (doRelu) { v0 = fmaxf(v0, 0.0f); v1 = fmaxf(v1, 0.0f); }
    C[(size_t)row * ldc + col0] = v0;
    C[(size_t)row * ldc + col1] = v1;
    if (C16) {
      C16[(size_t)row * ldc + col0] = (half_t)v0;
      C16[(size_t)row * ldc + col1] = (half_t)v1;
    }
  }
}

// ---------------------------------------------------------------------------
// Attention scores: s[n,h] = sum_f xp[n,h*32+f]*as[h,f]; d likewise
// ---------------------------------------------------------------------------
__global__ void sd_kernel(const float* __restrict__ xp,
                          const float* __restrict__ a_s,
                          const float* __restrict__ a_d,
                          float* __restrict__ s, float* __restrict__ d) {
  int t = blockIdx.x * blockDim.x + threadIdx.x;   // < 4*1536*4
  if (t >= NB * NN * NH) return;
  int h = t & 3;
  int node = (t >> 2) % NN;
  int b = (t >> 2) / NN;
  const float* row = xp + ((size_t)(b * NN + node) << 7) + h * OF;
  float ss = 0.f, dd = 0.f;
#pragma unroll 8
  for (int f = 0; f < OF; ++f) {
    float v = row[f];
    ss += v * a_s[h * OF + f];
    dd += v * a_d[h * OF + f];
  }
  s[t] = ss;
  d[t] = dd;
}

// ---------------------------------------------------------------------------
// Softmax stats per (b, dst node, head): one wave scans all in-edges.
// ---------------------------------------------------------------------------
__global__ void stat_kernel(const float* __restrict__ s,
                            const float* __restrict__ d,
                            float* __restrict__ mx, float* __restrict__ idn) {
  int tid = blockIdx.x * blockDim.x + threadIdx.x;
  int w = tid >> 5;
  int lane = threadIdx.x & 31;
  int h = w & 3;
  int node = (w >> 2) % NN;
  int b = (w >> 2) / NN;
  if (b >= NB) return;
  const float* sb = s + (size_t)b * NN * NH;
  size_t id = ((size_t)b * NN + node) * NH + h;
  float dv = d[id];
  int nsrc = (node < NSEC) ? (NPRIM + 1)
           : (node < NSEC + NPRIM) ? (NSEC + 2) : 2;
  float m = -1e30f;
  for (int idx = lane; idx < nsrc; idx += 32) {
    int srcn;
    if (node < NSEC)            srcn = (idx < NPRIM) ? (NSEC + idx) : node;
    else if (node < NSEC+NPRIM) srcn = (idx < NSEC) ? idx
                                       : (idx == NSEC ? node + NPRIM : node);
    else                        srcn = (idx == 0) ? (node - NPRIM) : node;
    m = fmaxf(m, lrelu(sb[srcn * NH + h] + dv));
  }
#pragma unroll
  for (int o = 16; o >= 1; o >>= 1) m = fmaxf(m, __shfl_xor(m, o, 32));
  float sum = 0.f;
  for (int idx = lane; idx < nsrc; idx += 32) {
    int srcn;
    if (node < NSEC)            srcn = (idx < NPRIM) ? (NSEC + idx) : node;
    else if (node < NSEC+NPRIM) srcn = (idx < NSEC) ? idx
                                       : (idx == NSEC ? node + NPRIM : node);
    else                        srcn = (idx == 0) ? (node - NPRIM) : node;
    sum += __expf(lrelu(sb[srcn * NH + h] + dv) - m);
  }
#pragma unroll
  for (int o = 16; o >= 1; o >>= 1) sum += __shfl_xor(sum, o, 32);
  if (lane == 0) { mx[id] = m; idn[id] = 1.0f / sum; }
}

// ---------------------------------------------------------------------------
// alpha matrices precomputed in f16 directly in the WMMA A-operand layout:
//   lane<16: M=lane, K = hi*16 + j with hi=lane>>4 (contiguous K order used here, as in xpT).
// aB: [b][h][jt=16][kt=32][lane][j]  (beam-dst over 1024 sec sources)
// aS: [b][h][it=64][kt=8 ][lane][j]  (sec-dst  over  256 beam sources)
// ---------------------------------------------------------------------------
__global__ void alpha_beam_kernel(const float* __restrict__ s,
                                  const float* __restrict__ d,
                                  const float* __restrict__ mx,
                                  const float* __restrict__ idn,
                                  half_t* __restrict__ aB) {
  int t = blockIdx.x * blockDim.x + threadIdx.x;   // < 4M
  if (t >= (NB << 20)) return;
  int j = t & 15, lane = (t >> 4) & 31, kt = (t >> 9) & 31;
  int jt = (t >> 14) & 15, h = (t >> 18) & 3, b = t >> 20;
  int ml = lane & 15, hi = lane >> 4;
  int klocal = hi * 16 + j;
  int src = kt * 32 + klocal;                 // sec node (0..1023)
  int dst = NSEC + jt * 16 + ml;              // beam node
  size_t is = ((size_t)b * NN + src) * NH + h;
  size_t id = ((size_t)b * NN + dst) * NH + h;
  float al = __expf(lrelu(s[is] + d[id]) - mx[id]) * idn[id];
  aB[t] = (half_t)al;
}

__global__ void alpha_sec_kernel(const float* __restrict__ s,
                                 const float* __restrict__ d,
                                 const float* __restrict__ mx,
                                 const float* __restrict__ idn,
                                 half_t* __restrict__ aS) {
  int t = blockIdx.x * blockDim.x + threadIdx.x;   // < 4M
  if (t >= (NB << 20)) return;
  int j = t & 15, lane = (t >> 4) & 31, kt = (t >> 9) & 7;
  int it = (t >> 12) & 63, h = (t >> 18) & 3, b = t >> 20;
  int ml = lane & 15, hi = lane >> 4;
  int klocal = hi * 16 + j;
  int src = NSEC + kt * 32 + klocal;          // beam node
  int dst = it * 16 + ml;                     // sec node
  size_t is = ((size_t)b * NN + src) * NH + h;
  size_t id = ((size_t)b * NN + dst) * NH + h;
  float al = __expf(lrelu(s[is] + d[id]) - mx[id]) * idn[id];
  aS[t] = (half_t)al;
}

// ---------------------------------------------------------------------------
// WMMA aggregation: beam-dst  y[beam,hF] = alpha[16x1024] @ xp_sec[1024x32]
// One wave per (b, head, beam 16-tile); two 16-wide N tiles per head
// (independent accumulators, interleaved WMMAs).
// ---------------------------------------------------------------------------
__global__ void agg_beam_kernel(const half_t* __restrict__ aB,
                                const half_t* __restrict__ xpT,
                                float* __restrict__ y) {
  int tid = blockIdx.x * blockDim.x + threadIdx.x;
  int w = tid >> 5;                 // 0..255
  int lane = threadIdx.x & 31;
  int jt = w & 15, h = (w >> 4) & 3, b = w >> 6;
  int ml = lane & 15, hi = lane >> 4;
  const half_t* xT = xpT + (size_t)b * FG * NN;
  const half_t* Abase = aB + (size_t)((b * NH + h) * 16 + jt) * 32 * 512;
  const half_t* brow0 = xT + (size_t)(h * OF + ml) * NN;
  const half_t* brow1 = xT + (size_t)(h * OF + 16 + ml) * NN;
  v8f acc0 = {0.f, 0.f, 0.f, 0.f, 0.f, 0.f, 0.f, 0.f};
  v8f acc1 = {0.f, 0.f, 0.f, 0.f, 0.f, 0.f, 0.f, 0.f};
  for (int kt = 0; kt < 32; ++kt) {
    v16h a = *(const v16h*)(Abase + (size_t)kt * 512 + lane * 16);
    int kb = kt * 32 + hi * 16;               // src node base (sec)
    v16h b0 = *(const v16h*)(brow0 + kb);
    v16h b1 = *(const v16h*)(brow1 + kb);
    acc0 = __builtin_amdgcn_wmma_f32_16x16x32_f16(false, a, false, b0,
                                                  (short)0, acc0, false, false);
    acc1 = __builtin_amdgcn_wmma_f32_16x16x32_f16(false, a, false, b1,
                                                  (short)0, acc1, false, false);
  }
#pragma unroll
  for (int r = 0; r < 8; ++r) {
    int node = NSEC + jt * 16 + r + hi * 8;
    float* yr = y + ((size_t)(b * NN + node) << 7) + h * OF;
    yr[ml] = acc0[r];
    yr[16 + ml] = acc1[r];
  }
}

// sec-dst: y[sec,hF] = alpha[16x256] @ xp_beam[256x32]
__global__ void agg_sec_kernel(const half_t* __restrict__ aS,
                               const half_t* __restrict__ xpT,
                               float* __restrict__ y) {
  int tid = blockIdx.x * blockDim.x + threadIdx.x;
  int w = tid >> 5;                 // 0..1023
  int lane = threadIdx.x & 31;
  int it = w & 63, h = (w >> 6) & 3, b = w >> 8;
  int ml = lane & 15, hi = lane >> 4;
  const half_t* xT = xpT + (size_t)b * FG * NN;
  const half_t* Abase = aS + (size_t)((b * NH + h) * 64 + it) * 8 * 512;
  const half_t* brow0 = xT + (size_t)(h * OF + ml) * NN;
  const half_t* brow1 = xT + (size_t)(h * OF + 16 + ml) * NN;
  v8f acc0 = {0.f, 0.f, 0.f, 0.f, 0.f, 0.f, 0.f, 0.f};
  v8f acc1 = {0.f, 0.f, 0.f, 0.f, 0.f, 0.f, 0.f, 0.f};
  for (int kt = 0; kt < 8; ++kt) {
    v16h a = *(const v16h*)(Abase + (size_t)kt * 512 + lane * 16);
    int kb = NSEC + kt * 32 + hi * 16;        // src node base (beam)
    v16h b0 = *(const v16h*)(brow0 + kb);
    v16h b1 = *(const v16h*)(brow1 + kb);
    acc0 = __builtin_amdgcn_wmma_f32_16x16x32_f16(false, a, false, b0,
                                                  (short)0, acc0, false, false);
    acc1 = __builtin_amdgcn_wmma_f32_16x16x32_f16(false, a, false, b1,
                                                  (short)0, acc1, false, false);
  }
#pragma unroll
  for (int r = 0; r < 8; ++r) {
    int node = it * 16 + r + hi * 8;
    float* yr = y + ((size_t)(b * NN + node) << 7) + h * OF;
    yr[ml] = acc0[r];
    yr[16 + ml] = acc1[r];
  }
}

// ---------------------------------------------------------------------------
// Post pass: self-loop (+prim/beam pair) contributions, bias, ReLU.
// ---------------------------------------------------------------------------
__global__ void post_kernel(const float* __restrict__ xp,
                            const float* __restrict__ s,
                            const float* __restrict__ d,
                            const float* __restrict__ mx,
                            const float* __restrict__ idn,
                            const float* __restrict__ bias,
                            float* __restrict__ y) {
  int t = blockIdx.x * blockDim.x + threadIdx.x;   // < 786432
  if (t >= NB * NN * FG) return;
  int f = t & 127;
  int node = (t >> 7) % NN;
  int b = (t >> 7) / NN;
  int h = f >> 5;
  size_t nb = (size_t)b * NN;
  size_t id = (nb + node) * NH + h;
  float dv = d[id], mv = mx[id], iv = idn[id];
  const float* sb = s + nb * NH;
  const float* xb = xp + (nb << 7);
  auto alpha = [&](int srcn) {
    return __expf(lrelu(sb[srcn * NH + h] + dv) - mv) * iv;
  };
  size_t yi = ((nb + node) << 7) + f;
  float acc;
  if (node < NSEC) {
    acc = y[yi] + alpha(node) * xb[((size_t)node << 7) + f];
  } else if (node < NSEC + NPRIM) {
    int pn = node + NPRIM;
    acc = y[yi] + alpha(node) * xb[((size_t)node << 7) + f]
                + alpha(pn) * xb[((size_t)pn << 7) + f];
  } else {
    int bn = node - NPRIM;
    acc = alpha(bn) * xb[((size_t)bn << 7) + f]
        + alpha(node) * xb[((size_t)node << 7) + f];
  }
  acc += bias[f];
  y[yi] = fmaxf(acc, 0.f);
}

// ---------------------------------------------------------------------------
// Final rescale chain
// ---------------------------------------------------------------------------
__global__ void zero4_kernel(float* __restrict__ p) {
  if (threadIdx.x < NB) p[threadIdx.x] = 0.f;
}

__global__ void psum_kernel(const float* __restrict__ p,
                            float* __restrict__ psum) {
  __shared__ float sd[256];
  int b = blockIdx.x >> 8;
  int chunk = blockIdx.x & 255;
  size_t base = (size_t)b * (NSEC * FC1D) + (size_t)chunk * 1024;
  float v = 0.f;
#pragma unroll
  for (int i = 0; i < 4; ++i) v += p[base + threadIdx.x + i * 256];
  sd[threadIdx.x] = v;
  __syncthreads();
  for (int st = 128; st > 0; st >>= 1) {
    if (threadIdx.x < st) sd[threadIdx.x] += sd[threadIdx.x + st];
    __syncthreads();
  }
  if (threadIdx.x == 0) atomicAdd(&psum[b], sd[0]);
}

__global__ void ratio_kernel(const float* __restrict__ pP,
                             const float* __restrict__ pT,
                             const float* __restrict__ psum,
                             float* __restrict__ ratio) {
  int b = threadIdx.x;
  if (b < NB) {
    float sp = 0.f;
    for (int i = 0; i < NPRIM; ++i) sp += pP[b * NPRIM + i];
    float rem = fmaxf(pT[0] - sp, 0.f);
    float r = rem / (psum[b] + 1e-5f);
    ratio[b] = fminf(r, 1.0f);
  }
}

__global__ void final_kernel(const float* __restrict__ p,
                             const float* __restrict__ ratio,
                             float* __restrict__ out) {
  int t = blockIdx.x * blockDim.x + threadIdx.x;   // < 1048576
  if (t >= NB * NSEC * FC1D) return;
  int b = t >> 18;
  float v = p[t] * ratio[b];
  out[t] = (v > 0.001f) ? v : 0.f;
}

// ---------------------------------------------------------------------------
// Host driver
// ---------------------------------------------------------------------------
static inline size_t alignup(size_t x) { return (x + 255) & ~(size_t)255; }

extern "C" void kernel_launch(void* const* d_in, const int* in_sizes, int n_in,
                              void* d_out, int out_size, void* d_ws, size_t ws_size,
                              hipStream_t stream) {
  (void)in_sizes; (void)n_in; (void)out_size; (void)ws_size;
  const float* h_in = (const float*)d_in[0];
  const float* pP   = (const float*)d_in[1];
  const float* pT   = (const float*)d_in[2];
  const float* W0   = (const float*)d_in[3];
  const float* b0   = (const float*)d_in[4];
  const float* as0  = (const float*)d_in[5];
  const float* ad0  = (const float*)d_in[6];
  const float* W1   = (const float*)d_in[7];
  const float* b1   = (const float*)d_in[8];
  const float* as1  = (const float*)d_in[9];
  const float* ad1  = (const float*)d_in[10];
  const float* Wf0  = (const float*)d_in[11];
  const float* bf0  = (const float*)d_in[12];
  const float* Wf1  = (const float*)d_in[13];
  const float* bf1  = (const float*)d_in[14];
  float* out = (float*)d_out;

  uint8_t* ws = (uint8_t*)d_ws;
  size_t off = 0;
  auto carve = [&](size_t bytes) { void* p = ws + off; off += alignup(bytes); return p; };

  half_t* x16  = (half_t*)carve((size_t)NB * NN * FG * 2);     // also reused as hg16
  float*  xp   = (float*) carve((size_t)NB * NN * FG * 4);
  half_t* xpT  = (half_t*)carve((size_t)NB * FG * NN * 2);
  float*  sbuf = (float*) carve((size_t)NB * NN * NH * 4);
  float*  dbuf = (float*) carve((size_t)NB * NN * NH * 4);
  float*  mx   = (float*) carve((size_t)NB * NN * NH * 4);
  float*  idn  = (float*) carve((size_t)NB * NN * NH * 4);
  half_t* aB   = (half_t*)carve((size_t)(NB << 20) * 2);
  half_t* aS   = (half_t*)carve((size_t)(NB << 20) * 2);
  float*  y    = (float*) carve((size_t)NB * NN * FG * 4);
  half_t* w0h  = (half_t*)carve((size_t)FIN * FG * 2);
  half_t* w1h  = (half_t*)carve((size_t)FG * FG * 2);
  half_t* wf0h = (half_t*)carve((size_t)FG * FC0D * 2);
  half_t* wf1h = (half_t*)carve((size_t)FC0D * FC1D * 2);
  float*  z    = (float*) carve((size_t)NB * NSEC * FC0D * 4);
  half_t* z16  = (half_t*)carve((size_t)NB * NSEC * FC0D * 2);
  float*  p    = (float*) carve((size_t)NB * NSEC * FC1D * 4);
  float*  psum = (float*) carve(NB * 4);
  float*  ratio= (float*) carve(NB * 4);

  // ---- weight swizzles (B operand layout) ----
  swz_weight_kernel<<<(FIN * FG + 255) / 256, 256, 0, stream>>>(W0, w0h, 4, 8);
  swz_weight_kernel<<<(FG * FG + 255) / 256, 256, 0, stream>>>(W1, w1h, 4, 8);
  swz_weight_kernel<<<(FG * FC0D + 255) / 256, 256, 0, stream>>>(Wf0, wf0h, 4, 16);
  swz_weight_kernel<<<(FC0D * FC1D + 255) / 256, 256, 0, stream>>>(Wf1, wf1h, 8, 16);

  const int nAct = NB * NN * FG;          // 786432
  // proj: Mtiles=384, NT=8 -> N-tile pairs=4 -> 1536 waves -> 384 blocks of 128
  const int gemmProjBlocks = (NB * NN / 16) * 4 * 32 / 128;
  for (int layer = 0; layer < 2; ++layer) {
    const float* xin = (layer == 0) ? h_in : y;
    const half_t* wh = (layer == 0) ? w0h : w1h;
    const float* av_s = (layer == 0) ? as0 : as1;
    const float* av_d = (layer == 0) ? ad0 : ad1;
    const float* bias = (layer == 0) ? b0 : b1;

    cvt_f32_to_f16_kernel<<<nAct / 256, 256, 0, stream>>>(xin, x16, nAct);
    // xp = x @ W   (M=6144, K=128, N=128)
    gemm_ws_kernel<<<gemmProjBlocks, 128, 0, stream>>>(
        x16, FG, wh, nullptr, xp, nullptr, FG, NB * NN / 16, 4, 8, 0);
    xpT_kernel<<<nAct / 256, 256, 0, stream>>>(xp, xpT);
    sd_kernel<<<(NB * NN * NH + 255) / 256, 256, 0, stream>>>(xp, av_s, av_d, sbuf, dbuf);
    stat_kernel<<<NB * NN * NH * 32 / 128, 128, 0, stream>>>(sbuf, dbuf, mx, idn);
    alpha_beam_kernel<<<(NB << 20) / 256, 256, 0, stream>>>(sbuf, dbuf, mx, idn, aB);
    alpha_sec_kernel<<<(NB << 20) / 256, 256, 0, stream>>>(sbuf, dbuf, mx, idn, aS);
    agg_beam_kernel<<<NB * NH * 16 * 32 / 128, 128, 0, stream>>>(aB, xpT, y);
    agg_sec_kernel<<<NB * NH * 64 * 32 / 128, 128, 0, stream>>>(aS, xpT, y);
    post_kernel<<<nAct / 256, 256, 0, stream>>>(xp, sbuf, dbuf, mx, idn, bias, y);
  }

  // ---- FC stack on sec nodes ----
  cvt_sec_kernel<<<(NB * NSEC * FG) / 256, 256, 0, stream>>>(y, x16);  // hg16
  // z = relu(hg @ Wf0 + bf0)   (M=4096, K=128, N=256): 256*8=2048 waves
  gemm_ws_kernel<<<(NB * NSEC / 16) * 8 * 32 / 128, 128, 0, stream>>>(
      x16, FG, wf0h, bf0, z, z16, FC0D, NB * NSEC / 16, 4, 16, 1);
  // p = relu(z @ Wf1 + bf1)    (M=4096, K=256, N=256)
  gemm_ws_kernel<<<(NB * NSEC / 16) * 8 * 32 / 128, 128, 0, stream>>>(
      z16, FC0D, wf1h, bf1, p, nullptr, FC1D, NB * NSEC / 16, 8, 16, 1);

  // ---- power rescale + threshold ----
  zero4_kernel<<<1, 32, 0, stream>>>(psum);
  psum_kernel<<<NB * 256, 256, 0, stream>>>(p, psum);
  ratio_kernel<<<1, 32, 0, stream>>>(pP, pT, psum, ratio);
  final_kernel<<<(NB * NSEC * FC1D) / 256, 256, 0, stream>>>(p, ratio, out);
}
